// Transcoder_82566451298892
// MI455X (gfx1250) — compile-verified, matrix operand order checked by reading, not run on hardware
//
#include <hip/hip_runtime.h>
#include <hip/hip_bf16.h>

typedef __attribute__((ext_vector_type(16))) __bf16 v16bf;
typedef __attribute__((ext_vector_type(8)))  __bf16 v8bf;
typedef __attribute__((ext_vector_type(8)))  float  v8f;

#define N_TOK 8192
#define D_IN  1024
#define D_H   16384
#define D_OUT 1024
#define TOPK  64

#define WMMA_BF16(A, B, C) \
  __builtin_amdgcn_wmma_f32_16x16x32_bf16(false, (A), false, (B), (short)0, (C), false, false)

// ---------------- fp32 -> bf16 (RNE), 4-wide ----------------
__global__ void convert_f32_bf16(const float4* __restrict__ src,
                                 ushort4* __restrict__ dst, int n4) {
  int i = blockIdx.x * blockDim.x + threadIdx.x;
  if (i < n4) {
    float4 f = src[i];
    ushort4 o;
    unsigned u;
    u = __float_as_uint(f.x); o.x = (unsigned short)((u + 0x7FFFu + ((u >> 16) & 1u)) >> 16);
    u = __float_as_uint(f.y); o.y = (unsigned short)((u + 0x7FFFu + ((u >> 16) & 1u)) >> 16);
    u = __float_as_uint(f.z); o.z = (unsigned short)((u + 0x7FFFu + ((u >> 16) & 1u)) >> 16);
    u = __float_as_uint(f.w); o.w = (unsigned short)((u + 0x7FFFu + ((u >> 16) & 1u)) >> 16);
    dst[i] = o;
  }
}

// ---------------- W2 [D_OUT, D_H] -> W2T [D_H, D_OUT] ----------------
__global__ __launch_bounds__(256) void transpose_w2(const float* __restrict__ W2,
                                                    float* __restrict__ W2T) {
  __shared__ float tile[32][33];
  int j0 = blockIdx.x * 32;     // along D_H
  int o0 = blockIdx.y * 32;     // along D_OUT
  for (int r = threadIdx.y; r < 32; r += 8)
    tile[r][threadIdx.x] = W2[(size_t)(o0 + r) * D_H + j0 + threadIdx.x];
  __syncthreads();
  for (int r = threadIdx.y; r < 32; r += 8)
    W2T[(size_t)(j0 + r) * D_OUT + o0 + threadIdx.x] = tile[threadIdx.x][r];
}

// ---------------- GEMM1: h = x @ W1^T + b1 (bf16 WMMA, f32 acc) ----------------
// block = 128 threads (4 waves). Wave tile: 32M x 64N (2 A-frags x 4 B-frags ->
// 8 WMMAs per 32-k step). Register double-buffered K pipeline: next step's 12
// fragment loads are in flight while the current step's 8 WMMAs execute.
__device__ __forceinline__ v16bf load_frag(const __bf16* __restrict__ p) {
  union { v16bf v; v8bf q[2]; } f;
  f.q[0] = *(const v8bf*)(p);
  f.q[1] = *(const v8bf*)(p + 16);
  return f.v;
}

__global__ __launch_bounds__(128) void gemm1_wmma(
    const unsigned short* __restrict__ xb,   // [N_TOK, D_IN] bf16
    const unsigned short* __restrict__ w1b,  // [D_H, D_IN]  bf16
    const float* __restrict__ b1,            // [D_H]
    float* __restrict__ h)                   // [N_TOK, D_H]
{
  const int wave  = threadIdx.x >> 5;
  const int lane  = threadIdx.x & 31;
  const int lhalf = lane >> 4;   // 0|1
  const int l16   = lane & 15;
  const int nb    = blockIdx.x * 256 + wave * 64;  // wave's N base
  const int mb    = blockIdx.y * 32;               // block's M base
  const int klo   = lhalf * 8;   // K sub-chunk select per ISA 16-bit A/B layout

  const __bf16* xa = (const __bf16*)xb;
  const __bf16* wb = (const __bf16*)w1b;

  const __bf16* pa0 = xa + (size_t)(mb +  0 + l16) * D_IN + klo;
  const __bf16* pa1 = xa + (size_t)(mb + 16 + l16) * D_IN + klo;
  const __bf16* pb0 = wb + (size_t)(nb +  0 + l16) * D_IN + klo;
  const __bf16* pb1 = wb + (size_t)(nb + 16 + l16) * D_IN + klo;
  const __bf16* pb2 = wb + (size_t)(nb + 32 + l16) * D_IN + klo;
  const __bf16* pb3 = wb + (size_t)(nb + 48 + l16) * D_IN + klo;

  v8f a00 = {}, a01 = {}, a02 = {}, a03 = {};
  v8f a10 = {}, a11 = {}, a12 = {}, a13 = {};

  // prologue: stage k-step 0
  v16bf cA0 = load_frag(pa0), cA1 = load_frag(pa1);
  v16bf cB0 = load_frag(pb0), cB1 = load_frag(pb1);
  v16bf cB2 = load_frag(pb2), cB3 = load_frag(pb3);

  for (int kb = 0; kb < D_IN; kb += 64) {
    const int k1 = kb + 32;
    // stage k-step kb+32 while computing kb
    v16bf nA0 = load_frag(pa0 + k1), nA1 = load_frag(pa1 + k1);
    v16bf nB0 = load_frag(pb0 + k1), nB1 = load_frag(pb1 + k1);
    v16bf nB2 = load_frag(pb2 + k1), nB3 = load_frag(pb3 + k1);

    a00 = WMMA_BF16(cA0, cB0, a00);
    a01 = WMMA_BF16(cA0, cB1, a01);
    a02 = WMMA_BF16(cA0, cB2, a02);
    a03 = WMMA_BF16(cA0, cB3, a03);
    a10 = WMMA_BF16(cA1, cB0, a10);
    a11 = WMMA_BF16(cA1, cB1, a11);
    a12 = WMMA_BF16(cA1, cB2, a12);
    a13 = WMMA_BF16(cA1, cB3, a13);

    const int k2 = kb + 64;
    if (k2 < D_IN) {
      // stage k-step kb+64 while computing kb+32
      cA0 = load_frag(pa0 + k2); cA1 = load_frag(pa1 + k2);
      cB0 = load_frag(pb0 + k2); cB1 = load_frag(pb1 + k2);
      cB2 = load_frag(pb2 + k2); cB3 = load_frag(pb3 + k2);
    }

    a00 = WMMA_BF16(nA0, nB0, a00);
    a01 = WMMA_BF16(nA0, nB1, a01);
    a02 = WMMA_BF16(nA0, nB2, a02);
    a03 = WMMA_BF16(nA0, nB3, a03);
    a10 = WMMA_BF16(nA1, nB0, a10);
    a11 = WMMA_BF16(nA1, nB1, a11);
    a12 = WMMA_BF16(nA1, nB2, a12);
    a13 = WMMA_BF16(nA1, nB3, a13);
  }

  // C/D layout: lane -> N = lane%16; VGPR v -> M = v + 8*(lane/16)
  const float bias0 = b1[nb +  0 + l16];
  const float bias1 = b1[nb + 16 + l16];
  const float bias2 = b1[nb + 32 + l16];
  const float bias3 = b1[nb + 48 + l16];
#pragma unroll
  for (int v = 0; v < 8; ++v) {
    const size_t r0 = (size_t)(mb +  0 + v + 8 * lhalf) * D_H + nb + l16;
    const size_t r1 = (size_t)(mb + 16 + v + 8 * lhalf) * D_H + nb + l16;
    h[r0 +  0] = a00[v] + bias0;
    h[r0 + 16] = a01[v] + bias1;
    h[r0 + 32] = a02[v] + bias2;
    h[r0 + 48] = a03[v] + bias3;
    h[r1 +  0] = a10[v] + bias0;
    h[r1 + 16] = a11[v] + bias1;
    h[r1 + 32] = a12[v] + bias2;
    h[r1 + 48] = a13[v] + bias3;
  }
}

// ---------------- exact top-K per row via 4-pass radix select ----------------
__device__ __forceinline__ unsigned f2key(float f) {
  unsigned u = __float_as_uint(f);
  return u ^ ((unsigned)((int)u >> 31) | 0x80000000u);  // ascending-monotonic
}

__global__ __launch_bounds__(256) void topk_select(
    const float* __restrict__ h, int* __restrict__ oidx, float* __restrict__ oval)
{
  const int n = blockIdx.x;
  const float* row = h + (size_t)n * D_H;
  const int tid = threadIdx.x;

  __shared__ unsigned hist[256];
  __shared__ unsigned sh_sel, sh_rem, sh_cnt;

  unsigned prefix = 0;
  unsigned rem = TOPK;
  for (int p = 0; p < 4; ++p) {
    const int shift = 24 - 8 * p;
    hist[tid] = 0;
    __syncthreads();
    for (int i = tid; i < D_H; i += 256) {
      unsigned key = f2key(row[i]);
      if (p == 0 || (key >> (shift + 8)) == prefix)
        atomicAdd(&hist[(key >> shift) & 255u], 1u);
    }
    __syncthreads();
    if (tid == 0) {
      unsigned acc = 0, sel = 0, r = rem;
      for (int b = 255; b >= 0; --b) {
        acc += hist[b];
        if (acc >= r) { sel = (unsigned)b; r -= (acc - hist[b]); break; }
      }
      sh_sel = sel; sh_rem = r;
    }
    __syncthreads();
    prefix = (prefix << 8) | sh_sel;
    rem = sh_rem;
  }

  const unsigned T = prefix;  // key of the K-th largest
  if (tid == 0) sh_cnt = 0;
  __syncthreads();
  // strictly greater than threshold
  for (int i = tid; i < D_H; i += 256) {
    float v = row[i];
    if (f2key(v) > T) {
      unsigned pos = atomicAdd(&sh_cnt, 1u);
      oidx[(size_t)n * TOPK + pos] = i;
      oval[(size_t)n * TOPK + pos] = v;
    }
  }
  __syncthreads();
  // fill remaining slots with ties at threshold
  for (int i = tid; i < D_H; i += 256) {
    float v = row[i];
    if (f2key(v) == T) {
      unsigned pos = atomicAdd(&sh_cnt, 1u);
      if (pos < TOPK) {
        oidx[(size_t)n * TOPK + pos] = i;
        oval[(size_t)n * TOPK + pos] = v;
      }
    }
  }
}

// ---------------- GEMM2: out = hs @ W2^T + b2 (sparse gather over W2T rows) ----
__global__ __launch_bounds__(256) void gemm2_scatter(
    const int* __restrict__ oidx, const float* __restrict__ oval,
    const float* __restrict__ W2T, const float* __restrict__ b2,
    float* __restrict__ out)
{
  const int n = blockIdx.x;
  const int tid = threadIdx.x;
  __shared__ float sval[TOPK];
  __shared__ int   sidx[TOPK];
  if (tid < TOPK) {
    sidx[tid] = oidx[(size_t)n * TOPK + tid];
    sval[tid] = oval[(size_t)n * TOPK + tid];
  }
  __syncthreads();
  float a0 = b2[tid], a1 = b2[tid + 256], a2 = b2[tid + 512], a3 = b2[tid + 768];
#pragma unroll 4
  for (int k = 0; k < TOPK; ++k) {
    const float v = sval[k];
    const float* w = W2T + (size_t)sidx[k] * D_OUT;
    a0 += v * w[tid];
    a1 += v * w[tid + 256];
    a2 += v * w[tid + 512];
    a3 += v * w[tid + 768];
  }
  float* orow = out + (size_t)n * D_OUT;
  orow[tid] = a0; orow[tid + 256] = a1; orow[tid + 512] = a2; orow[tid + 768] = a3;
}

extern "C" void kernel_launch(void* const* d_in, const int* in_sizes, int n_in,
                              void* d_out, int out_size, void* d_ws, size_t ws_size,
                              hipStream_t stream) {
  const float* x  = (const float*)d_in[0];   // [8192,1024]
  const float* W1 = (const float*)d_in[1];   // [16384,1024]
  const float* b1 = (const float*)d_in[2];   // [16384]
  const float* W2 = (const float*)d_in[3];   // [1024,16384]
  const float* b2 = (const float*)d_in[4];   // [1024]
  float* out = (float*)d_out;                // [8192,1024]

  // workspace carve-up (256B aligned)
  char* ws = (char*)d_ws;
  size_t off = 0;
  auto alloc = [&](size_t bytes) { char* p = ws + off; off = (off + bytes + 255) & ~(size_t)255; return p; };
  unsigned short* xb  = (unsigned short*)alloc((size_t)N_TOK * D_IN * 2);  // 16 MiB
  unsigned short* w1b = (unsigned short*)alloc((size_t)D_H * D_IN * 2);    // 32 MiB
  float* hbuf = (float*)alloc((size_t)N_TOK * D_H * 4);                    // 512 MiB
  float* W2T  = (float*)alloc((size_t)D_H * D_OUT * 4);                    // 64 MiB
  int*   idx  = (int*)alloc((size_t)N_TOK * TOPK * 4);                     // 2 MiB
  float* val  = (float*)alloc((size_t)N_TOK * TOPK * 4);                   // 2 MiB

  {
    int n4 = (N_TOK * D_IN) / 4;
    convert_f32_bf16<<<(n4 + 255) / 256, 256, 0, stream>>>((const float4*)x, (ushort4*)xb, n4);
  }
  {
    int n4 = (D_H * D_IN) / 4;
    convert_f32_bf16<<<(n4 + 255) / 256, 256, 0, stream>>>((const float4*)W1, (ushort4*)w1b, n4);
  }
  transpose_w2<<<dim3(D_H / 32, D_OUT / 32), dim3(32, 8), 0, stream>>>(W2, W2T);

  gemm1_wmma<<<dim3(D_H / 256, N_TOK / 32), 128, 0, stream>>>(xb, w1b, b1, hbuf);

  topk_select<<<N_TOK, 256, 0, stream>>>(hbuf, idx, val);

  gemm2_scatter<<<N_TOK, 256, 0, stream>>>(idx, val, W2T, b2, out);
}
